// EncoderLayer_81973745812020
// MI455X (gfx1250) — matrix operand order repaired and checked by reading, compiled, not run on hardware
//
#include <hip/hip_runtime.h>
#include <hip/hip_bf16.h>

// ---------------------------------------------------------------------------
// Problem constants (match the reference)
// ---------------------------------------------------------------------------
#define BB   4
#define SS   1024
#define HID  1024
#define NH   16
#define HD   64
#define PF   4096
#define MTOK (BB * SS)   // 4096 token rows

typedef __attribute__((ext_vector_type(16))) __bf16 bf16x16;
typedef __attribute__((ext_vector_type(8)))  float  floatx8;

union FragBF {
    bf16x16 v;
    uint4   q[2];
};

__device__ __forceinline__ floatx8 wmma_bf16(bf16x16 a, bf16x16 b, floatx8 c) {
    return __builtin_amdgcn_wmma_f32_16x16x32_bf16(
        /*neg_a=*/false, a, /*neg_b=*/false, b,
        /*c_mod=*/(short)0, c, /*reuse_a=*/false, /*reuse_b=*/false);
}

// A fragment (16x32, 16-bit): lane = (g, m16); holds k = g*16 .. g*16+15
// contiguous (32B), matching B -> two b128 loads.
__device__ __forceinline__ bf16x16 load_a_frag(const __bf16* rowptr, int g) {
    FragBF f;
    f.q[0] = *(const uint4*)(rowptr + g * 16);
    f.q[1] = *(const uint4*)(rowptr + g * 16 + 8);
    return f.v;
}

// B fragment (32x16, 16-bit): lane = (g, n16); holds k = g*16 .. g*16+15
// contiguous (32B) -> two b128 loads. colptr points at this lane's column, k=0.
__device__ __forceinline__ bf16x16 load_b_frag(const __bf16* colptr, int g) {
    FragBF f;
    f.q[0] = *(const uint4*)(colptr + g * 16);
    f.q[1] = *(const uint4*)(colptr + g * 16 + 8);
    return f.v;
}

// Async global -> LDS copy of 16B per lane (CDNA5 path, tracked by ASYNCcnt).
__device__ __forceinline__ void async_b128_to_lds(unsigned lds_off, const __bf16* gptr) {
    asm volatile("global_load_async_to_lds_b128 %0, %1, off"
                 :: "v"(lds_off), "v"(gptr) : "memory");
}

// ---------------------------------------------------------------------------
// f32 -> bf16 conversion
// ---------------------------------------------------------------------------
struct BH4 { __bf16 x, y, z, w; };

__global__ __launch_bounds__(256) void cvt_f32_bf16(const float* __restrict__ in,
                                                    __bf16* __restrict__ out, int n) {
    int i = blockIdx.x * 256 + threadIdx.x;
    if (i * 4 >= n) return;
    float4 f = ((const float4*)in)[i];
    BH4 o = { (__bf16)f.x, (__bf16)f.y, (__bf16)f.z, (__bf16)f.w };
    ((BH4*)out)[i] = o;
}

// ---------------------------------------------------------------------------
// Tiled WMMA GEMM: C[M,N] = A[M,K](bf16) @ W[N,K]^T(bf16) + bias
// Block tile 128x128, BK=32, 8 waves, wave tile 64x32 (4x2 WMMA subtiles).
// Global->LDS staging uses async-to-LDS DMA with double buffering; per-wave
// s_wait_asynccnt gates each tile before the block barrier.
// ---------------------------------------------------------------------------
#define EPI_F32       0
#define EPI_BF16RELU  1
#define EPI_HEADS     2   // bf16 out, [B,NH,S,HD] layout  (Q, K)
#define EPI_VT        3   // bf16 out, [B,NH,HD,S] layout  (V transposed)

template <int EPI>
__global__ __launch_bounds__(256) void gemm_bf16_wmma(
    const __bf16* __restrict__ A, const __bf16* __restrict__ W,
    const float* __restrict__ bias, float* __restrict__ outF,
    __bf16* __restrict__ outH, int Mdim, int Ndim, int Kdim) {

    constexpr int BM = 128, BN = 128, BK = 32, LDT = 40;  // 40 halves = 80B row stride
    __shared__ alignas(16) __bf16 As[2 * BM * LDT];
    __shared__ alignas(16) __bf16 Bs[2 * BN * LDT];

    const int tid  = threadIdx.x;
    const int lane = tid & 31;
    const int wid  = tid >> 5;
    const int g    = lane >> 4;
    const int n16  = lane & 15;

    const int bm = blockIdx.y * BM;
    const int bn = blockIdx.x * BN;

    // global->LDS copy assignment: 256 threads, each thread 32B of A and 32B of W
    const int lrow = tid >> 1;
    const int lseg = (tid & 1) * 16;
    const __bf16* gA = A + (size_t)(bm + lrow) * Kdim + lseg;
    const __bf16* gB = W + (size_t)(bn + lrow) * Kdim + lseg;

    // LDS byte offsets (flat shared address truncates to the LDS offset)
    const unsigned asb = (unsigned)(uintptr_t)(void*)&As[0];
    const unsigned bsb = (unsigned)(uintptr_t)(void*)&Bs[0];
    const unsigned thr_off = (unsigned)(lrow * LDT + lseg) * 2u;

    const int wm = (wid >> 2) * 64;  // wave row base in tile
    const int wn = (wid & 3) * 32;   // wave col base in tile

    floatx8 acc[4][2];
#pragma unroll
    for (int i = 0; i < 4; ++i)
#pragma unroll
        for (int j = 0; j < 2; ++j) acc[i][j] = floatx8{};

    auto issue_tile = [&](int k0, int buf) {
        const __bf16* pA = gA + k0;
        const __bf16* pB = gB + k0;
        const unsigned la = asb + (unsigned)(buf * BM * LDT) * 2u + thr_off;
        const unsigned lb = bsb + (unsigned)(buf * BN * LDT) * 2u + thr_off;
        async_b128_to_lds(la,      pA);
        async_b128_to_lds(la + 16, pA + 8);
        async_b128_to_lds(lb,      pB);
        async_b128_to_lds(lb + 16, pB + 8);
    };

    const int ntiles = Kdim / BK;
    issue_tile(0, 0);

    for (int t = 0; t < ntiles; ++t) {
        const int buf = t & 1;
        if (t + 1 < ntiles) {
            issue_tile((t + 1) * BK, buf ^ 1);
            // our 4 newest async ops may stay in flight; tile t is complete
            asm volatile("s_wait_asynccnt 4" ::: "memory");
        } else {
            asm volatile("s_wait_asynccnt 0" ::: "memory");
        }
        __syncthreads();  // every wave has waited for its own DMA -> tile ready

        const __bf16* Ab = &As[buf * BM * LDT];
        const __bf16* Bb = &Bs[buf * BN * LDT];

        bf16x16 af[4];
#pragma unroll
        for (int i = 0; i < 4; ++i)
            af[i] = load_a_frag(&Ab[(wm + i * 16 + n16) * LDT], g);
        bf16x16 bf_[2];
#pragma unroll
        for (int j = 0; j < 2; ++j)
            bf_[j] = load_b_frag(&Bb[(wn + j * 16 + n16) * LDT], g);

#pragma unroll
        for (int i = 0; i < 4; ++i)
#pragma unroll
            for (int j = 0; j < 2; ++j)
                acc[i][j] = wmma_bf16(af[i], bf_[j], acc[i][j]);

        __syncthreads();  // all reads done before this buffer is overwritten
    }

    // epilogue: C layout -> VGPR e: row = base + e + 8*g, col = base + n16
#pragma unroll
    for (int i = 0; i < 4; ++i) {
#pragma unroll
        for (int j = 0; j < 2; ++j) {
            const int col = bn + wn + j * 16 + n16;
            const float bval = bias[col];
#pragma unroll
            for (int e = 0; e < 8; ++e) {
                const int row = bm + wm + i * 16 + e + 8 * g;
                float v = acc[i][j][e] + bval;
                if constexpr (EPI == EPI_F32) {
                    outF[(size_t)row * Ndim + col] = v;
                } else if constexpr (EPI == EPI_BF16RELU) {
                    v = fmaxf(v, 0.0f);
                    outH[(size_t)row * Ndim + col] = (__bf16)v;
                } else if constexpr (EPI == EPI_HEADS) {
                    // [B,NH,S,HD]: (b*NH+h)*S*HD + s*HD + d
                    const int b_ = row >> 10, s_ = row & 1023;
                    const int h_ = col >> 6,  d_ = col & 63;
                    outH[((size_t)(b_ * NH + h_) << 16) + (s_ << 6) + d_] = (__bf16)v;
                } else {  // EPI_VT: [B,NH,HD,S]
                    const int b_ = row >> 10, s_ = row & 1023;
                    const int h_ = col >> 6,  d_ = col & 63;
                    outH[((size_t)((b_ * NH + h_) * HD + d_) << 10) + s_] = (__bf16)v;
                }
            }
        }
    }
}

// ---------------------------------------------------------------------------
// Flash attention: one wave per 16-row q tile; 8 waves (128 q rows) per block.
// Q,K: [B,NH,S,HD] bf16; Vt: [B,NH,HD,S] bf16; O: [B,S,HID] bf16.
// ---------------------------------------------------------------------------
__global__ __launch_bounds__(256) void flash_attn(
    const __bf16* __restrict__ Q, const __bf16* __restrict__ Km,
    const __bf16* __restrict__ Vt, const int* __restrict__ mask,
    __bf16* __restrict__ O) {

    constexpr int LDP = 40;
    __shared__ alignas(16) __bf16 Ps[8 * 16 * LDP];

    const int tid  = threadIdx.x;
    const int lane = tid & 31;
    const int wid  = tid >> 5;
    const int g    = lane >> 4;
    const int n16  = lane & 15;

    const int qblocks = SS / 128;             // 8
    const int qb = blockIdx.x % qblocks;
    const int bh = blockIdx.x / qblocks;      // b*NH + h
    const int h  = bh % NH;
    const int b  = bh / NH;
    const int qbase = qb * 128 + wid * 16;

    const __bf16* Qb = Q  + (size_t)bh * SS * HD;
    const __bf16* Kb = Km + (size_t)bh * SS * HD;
    const __bf16* Vb = Vt + (size_t)bh * HD * SS;
    const int*    mb = mask + b * SS;
    __bf16* Pw = &Ps[wid * 16 * LDP];

    // Q fragments (16x64 = two 16x32 A-frags)
    bf16x16 aq[2];
#pragma unroll
    for (int f = 0; f < 2; ++f) {
        const __bf16* qp = Qb + (size_t)(qbase + n16) * HD + f * 32;
        aq[f] = load_a_frag(qp, g);
    }

    float mrow[8], lrow[8];
#pragma unroll
    for (int r = 0; r < 8; ++r) { mrow[r] = -__builtin_inff(); lrow[r] = 0.0f; }
    floatx8 acc[4];
#pragma unroll
    for (int j = 0; j < 4; ++j) acc[j] = floatx8{};

    for (int kb = 0; kb < SS; kb += 32) {
        // S = Q @ K^T for two 16-key column subtiles
        floatx8 ev[2];
#pragma unroll
        for (int j = 0; j < 2; ++j) {
            const int key = kb + j * 16 + n16;
            const __bf16* kp = Kb + (size_t)key * HD;
            floatx8 z = floatx8{};
            z = wmma_bf16(aq[0], load_b_frag(kp, g), z);        // d = 0..31
            z = wmma_bf16(aq[1], load_b_frag(kp + 32, g), z);   // d = 32..63
            const float pen = (mb[key] == 0) ? -1e10f : 0.0f;
#pragma unroll
            for (int r = 0; r < 8; ++r) z[r] = z[r] * 0.125f + pen;  // 1/sqrt(64)
            ev[j] = z;
        }

        // online softmax: row stats via 16-lane-half xor reductions
        float alpha[8];
#pragma unroll
        for (int r = 0; r < 8; ++r) {
            float mx = fmaxf(ev[0][r], ev[1][r]);
            mx = fmaxf(mx, __shfl_xor(mx, 1, 32));
            mx = fmaxf(mx, __shfl_xor(mx, 2, 32));
            mx = fmaxf(mx, __shfl_xor(mx, 4, 32));
            mx = fmaxf(mx, __shfl_xor(mx, 8, 32));
            const float mn = fmaxf(mrow[r], mx);
            alpha[r] = __expf(mrow[r] - mn);
            mrow[r]  = mn;
            ev[0][r] = __expf(ev[0][r] - mn);
            ev[1][r] = __expf(ev[1][r] - mn);
            float sm = ev[0][r] + ev[1][r];
            sm += __shfl_xor(sm, 1, 32);
            sm += __shfl_xor(sm, 2, 32);
            sm += __shfl_xor(sm, 4, 32);
            sm += __shfl_xor(sm, 8, 32);
            lrow[r] = lrow[r] * alpha[r] + sm;
        }
#pragma unroll
        for (int j = 0; j < 4; ++j)
#pragma unroll
            for (int r = 0; r < 8; ++r) acc[j][r] *= alpha[r];

        // P (f32, C-layout) -> LDS bf16 [row][key] for A-fragment reload
#pragma unroll
        for (int j = 0; j < 2; ++j)
#pragma unroll
            for (int r = 0; r < 8; ++r)
                Pw[(r + 8 * g) * LDP + j * 16 + n16] = (__bf16)ev[j][r];
        asm volatile("s_wait_dscnt 0" ::: "memory");

        bf16x16 ap = load_a_frag(&Pw[n16 * LDP], g);

        // O += P @ V over 4 d-subtiles (Vt gives contiguous keys per lane)
#pragma unroll
        for (int j = 0; j < 4; ++j) {
            const int d = j * 16 + n16;
            const __bf16* vp = Vb + (size_t)d * SS + kb;
            acc[j] = wmma_bf16(ap, load_b_frag(vp, g), acc[j]);
        }
    }

    // normalize and store O as bf16 [B,S,HID]
#pragma unroll
    for (int r = 0; r < 8; ++r) {
        const float inv = 1.0f / lrow[r];
        const int row = qbase + r + 8 * g;
#pragma unroll
        for (int j = 0; j < 4; ++j) {
            const int col = h * HD + j * 16 + n16;
            O[((size_t)b * SS + row) * HID + col] = (__bf16)(acc[j][r] * inv);
        }
    }
}

// ---------------------------------------------------------------------------
// Residual + LayerNorm over HID=1024 cols; optional bf16 copy of the output.
// ---------------------------------------------------------------------------
__global__ __launch_bounds__(256) void ln_residual(
    const float* __restrict__ x, const float* __restrict__ resid,
    const float* __restrict__ gam, const float* __restrict__ bet,
    float* __restrict__ outF, __bf16* __restrict__ outH) {

    const int row = blockIdx.x;
    const int tid = threadIdx.x;
    const int lane = tid & 31, wid = tid >> 5;

    const float4 xv = ((const float4*)(x + (size_t)row * HID))[tid];
    const float4 rv = ((const float4*)(resid + (size_t)row * HID))[tid];
    float v[4] = { xv.x + rv.x, xv.y + rv.y, xv.z + rv.z, xv.w + rv.w };

    float s1 = v[0] + v[1] + v[2] + v[3];
    float s2 = v[0]*v[0] + v[1]*v[1] + v[2]*v[2] + v[3]*v[3];
#pragma unroll
    for (int mk = 1; mk < 32; mk <<= 1) {
        s1 += __shfl_xor(s1, mk, 32);
        s2 += __shfl_xor(s2, mk, 32);
    }
    __shared__ float r1[8], r2[8];
    if (lane == 0) { r1[wid] = s1; r2[wid] = s2; }
    __syncthreads();
    float t1 = 0.0f, t2 = 0.0f;
#pragma unroll
    for (int w = 0; w < 8; ++w) { t1 += r1[w]; t2 += r2[w]; }
    const float mean = t1 * (1.0f / HID);
    const float var  = t2 * (1.0f / HID) - mean * mean;
    const float rstd = rsqrtf(var + 1e-5f);

    const int cbase = tid * 4;
#pragma unroll
    for (int k = 0; k < 4; ++k) {
        const int c = cbase + k;
        const float y = (v[k] - mean) * rstd * gam[c] + bet[c];
        outF[(size_t)row * HID + c] = y;
        if (outH) outH[(size_t)row * HID + c] = (__bf16)y;
    }
}

// ---------------------------------------------------------------------------
// Host launcher
// ---------------------------------------------------------------------------
extern "C" void kernel_launch(void* const* d_in, const int* in_sizes, int n_in,
                              void* d_out, int out_size, void* d_ws, size_t ws_size,
                              hipStream_t stream) {
    (void)in_sizes; (void)n_in; (void)out_size; (void)ws_size;

    const float* src  = (const float*)d_in[0];
    const int*   mask = (const int*)d_in[1];
    const float* wq = (const float*)d_in[2];  const float* bq  = (const float*)d_in[3];
    const float* wk = (const float*)d_in[4];  const float* bk  = (const float*)d_in[5];
    const float* wv = (const float*)d_in[6];  const float* bv  = (const float*)d_in[7];
    const float* wo = (const float*)d_in[8];  const float* bo  = (const float*)d_in[9];
    const float* g1 = (const float*)d_in[10]; const float* be1 = (const float*)d_in[11];
    const float* w1 = (const float*)d_in[12]; const float* bf1 = (const float*)d_in[13];
    const float* w2 = (const float*)d_in[14]; const float* bf2 = (const float*)d_in[15];
    const float* g2 = (const float*)d_in[16]; const float* be2 = (const float*)d_in[17];
    float* out = (float*)d_out;

    char* p = (char*)d_ws;
    auto take = [&](size_t bytes) -> void* {
        void* r = (void*)p;
        p += (bytes + 255) & ~(size_t)255;
        return r;
    };
    __bf16* X16  = (__bf16*)take((size_t)MTOK * HID * 2);
    __bf16* Wq16 = (__bf16*)take((size_t)HID * HID * 2);
    __bf16* Wk16 = (__bf16*)take((size_t)HID * HID * 2);
    __bf16* Wv16 = (__bf16*)take((size_t)HID * HID * 2);
    __bf16* Wo16 = (__bf16*)take((size_t)HID * HID * 2);
    __bf16* W116 = (__bf16*)take((size_t)PF * HID * 2);
    __bf16* W216 = (__bf16*)take((size_t)HID * PF * 2);
    __bf16* Q16  = (__bf16*)take((size_t)MTOK * HID * 2);
    __bf16* K16  = (__bf16*)take((size_t)MTOK * HID * 2);
    __bf16* Vt16 = (__bf16*)take((size_t)MTOK * HID * 2);
    __bf16* O16  = (__bf16*)take((size_t)MTOK * HID * 2);
    float*  xo   = (float*)take((size_t)MTOK * HID * 4);
    float*  hbuf = (float*)take((size_t)MTOK * HID * 4);
    __bf16* h16  = (__bf16*)take((size_t)MTOK * HID * 2);
    __bf16* f1   = (__bf16*)take((size_t)MTOK * PF * 2);
    float*  f2   = (float*)take((size_t)MTOK * HID * 4);

    auto cvt = [&](const float* in, __bf16* o16, size_t n) {
        cvt_f32_bf16<<<(unsigned)(n / 4 / 256), 256, 0, stream>>>(in, o16, (int)n);
    };
    cvt(src, X16, (size_t)MTOK * HID);
    cvt(wq, Wq16, (size_t)HID * HID);
    cvt(wk, Wk16, (size_t)HID * HID);
    cvt(wv, Wv16, (size_t)HID * HID);
    cvt(wo, Wo16, (size_t)HID * HID);
    cvt(w1, W116, (size_t)PF * HID);
    cvt(w2, W216, (size_t)HID * PF);

    dim3 blk(256);
    dim3 g_proj(HID / 128, MTOK / 128);   // N=1024
    dim3 g_ffn1(PF / 128, MTOK / 128);    // N=4096
    dim3 g_ffn2(HID / 128, MTOK / 128);   // N=1024

    // Q, K in [B,NH,S,HD]; V transposed to [B,NH,HD,S]
    gemm_bf16_wmma<EPI_HEADS><<<g_proj, blk, 0, stream>>>(X16, Wq16, bq, nullptr, Q16,  MTOK, HID, HID);
    gemm_bf16_wmma<EPI_HEADS><<<g_proj, blk, 0, stream>>>(X16, Wk16, bk, nullptr, K16,  MTOK, HID, HID);
    gemm_bf16_wmma<EPI_VT>   <<<g_proj, blk, 0, stream>>>(X16, Wv16, bv, nullptr, Vt16, MTOK, HID, HID);

    // Attention (flash, no S x S materialization)
    flash_attn<<<BB * NH * (SS / 128), blk, 0, stream>>>(Q16, K16, Vt16, mask, O16);

    // Output projection, residual + LN1
    gemm_bf16_wmma<EPI_F32><<<g_proj, blk, 0, stream>>>(O16, Wo16, bo, xo, nullptr, MTOK, HID, HID);
    ln_residual<<<MTOK, blk, 0, stream>>>(xo, src, g1, be1, hbuf, h16);

    // FFN
    gemm_bf16_wmma<EPI_BF16RELU><<<g_ffn1, blk, 0, stream>>>(h16, W116, bf1, nullptr, f1, MTOK, PF, HID);
    gemm_bf16_wmma<EPI_F32>     <<<g_ffn2, blk, 0, stream>>>(f1,  W216, bf2, f2, nullptr, MTOK, HID, PF);

    // residual + LN2 -> final output
    ln_residual<<<MTOK, blk, 0, stream>>>(f2, hbuf, g2, be2, out, nullptr);
}
